// GPTDecoderBlock_11553462026829
// MI455X (gfx1250) — matrix operand order repaired and checked by reading, not compiled, not run on hardware
//
#include <hip/hip_runtime.h>
#include <math.h>

// ---------------------------------------------------------------------------
// Types for CDNA5 WMMA (wave32, 16x16x32 bf16 -> f32)
// ---------------------------------------------------------------------------
typedef __attribute__((ext_vector_type(16))) __bf16 v16bf;
typedef __attribute__((ext_vector_type(8)))  float  v8f;

union Frag16 {
    v16bf v;
    unsigned int u[8];
};

__device__ __forceinline__ v8f wmma_bf16(v16bf a, v16bf b, v8f c) {
    // D = A(16x32 bf16) x B(32x16 bf16) + C(16x16 f32)
    return __builtin_amdgcn_wmma_f32_16x16x32_bf16(
        false, a, false, b, (short)0, c, false, false);
}

__device__ __forceinline__ unsigned short f2bf(float f) {
    union { float f; unsigned int u; } c; c.f = f;
    unsigned int u = c.u;
    u += 0x7fffu + ((u >> 16) & 1u);   // round-to-nearest-even
    return (unsigned short)(u >> 16);
}

__device__ __forceinline__ float gelu_exact(float v) {
    return 0.5f * v * (1.0f + erff(v * 0.70710678118654752f));
}

// A-fragment (16x32 bf16) from a row-major tile of packed bf16 pairs.
// Per ISA: lane l (l<16) row M=l holds K in {0..7,16..23}; lane l+16 holds
// K in {8..15,24..31}.  dword index (K/2): j<4 -> j+hi*4 ; j>=4 -> j+4+hi*4.
__device__ __forceinline__ v16bf load_fragA(const unsigned int* tile, int row0,
                                            int strideDw, int kpair0, int lane) {
    const int r  = row0 + (lane & 15);
    const int off = ((lane >> 4) & 1) * 4;
    const unsigned int* p = tile + r * strideDw + kpair0;
    Frag16 f;
    f.u[0] = p[0 + off]; f.u[1] = p[1 + off]; f.u[2] = p[2 + off]; f.u[3] = p[3 + off];
    f.u[4] = p[8 + off]; f.u[5] = p[9 + off]; f.u[6] = p[10 + off]; f.u[7] = p[11 + off];
    return f.v;
}

// B-fragment (32x16 bf16) from a K-major tile: lane l holds column
// N = npair0*2 + (l&15); K order matches load_fragA (lane<16: {0..7,16..23},
// lane>=16: {8..15,24..31}).
__device__ __forceinline__ v16bf load_fragB(const unsigned int* tile, int row0,
                                            int strideDw, int npair0, int lane) {
    const unsigned short* t16 = (const unsigned short*)tile;
    const int n = npair0 * 2 + (lane & 15);
    const int hk = ((lane >> 4) & 1) * 8;
    Frag16 f;
#pragma unroll
    for (int j = 0; j < 8; ++j) {
        const int k = (j < 4) ? (hk + 2 * j) : (16 + hk + 2 * (j - 4));
        unsigned int lo = t16[(row0 + k) * strideDw * 2 + n];
        unsigned int up = t16[(row0 + k + 1) * strideDw * 2 + n];
        f.u[j] = lo | (up << 16);
    }
    return f.v;
}

// ---------------------------------------------------------------------------
// Elementwise helpers
// ---------------------------------------------------------------------------
__global__ void convert_mat_kernel(const float* __restrict__ src,
                                   unsigned short* __restrict__ dst,
                                   int total, int cols, int dstStride, int dstOff) {
    int idx = blockIdx.x * blockDim.x + threadIdx.x;
    if (idx < total) {
        int r = idx / cols, c = idx - r * cols;
        dst[(size_t)r * dstStride + dstOff + c] = f2bf(src[idx]);
    }
}

__global__ void concat_bias_kernel(const float* __restrict__ bq,
                                   const float* __restrict__ bk,
                                   const float* __restrict__ bv,
                                   float* __restrict__ o) {
    int i = blockIdx.x * blockDim.x + threadIdx.x;
    if (i < 768) { o[i] = bq[i]; o[768 + i] = bk[i]; o[1536 + i] = bv[i]; }
}

// One-shot K rearrange: ktg[((b*12+h)*64 + d) * S + s] = K[b,s,h,d]
// (coalesced writes over s; done once instead of per q-tile in flash attn).
__global__ __launch_bounds__(256) void k_transpose_kernel(
    const unsigned short* __restrict__ qkv, unsigned short* __restrict__ ktg, int S) {
    int idx = blockIdx.x * 256 + threadIdx.x;
    const int total = 24 * 64 * S;
    if (idx < total) {
        int s    = idx % S;
        int rest = idx / S;
        int d    = rest & 63;
        int bh   = rest >> 6;
        int b    = bh / 12, h = bh % 12;
        ktg[idx] = qkv[(size_t)(b * S + s) * 2304 + 768 + h * 64 + d];
    }
}

// LayerNorm over D=768, fp32 in -> bf16 out.  One block per row.
__global__ __launch_bounds__(256) void ln_bf16_kernel(
    const float* __restrict__ x, const float* __restrict__ gamma,
    const float* __restrict__ beta, unsigned short* __restrict__ out) {
    const int row = blockIdx.x;
    const float* xr = x + (size_t)row * 768;
    float s = 0.f, s2 = 0.f;
    for (int i = threadIdx.x; i < 768; i += 256) { float v = xr[i]; s += v; s2 += v * v; }
#pragma unroll
    for (int o = 16; o > 0; o >>= 1) { s += __shfl_down(s, o, 32); s2 += __shfl_down(s2, o, 32); }
    __shared__ float wsum[8], wsum2[8];
    __shared__ float mean_s, rstd_s;
    if ((threadIdx.x & 31) == 0) { wsum[threadIdx.x >> 5] = s; wsum2[threadIdx.x >> 5] = s2; }
    __syncthreads();
    if (threadIdx.x == 0) {
        float ts = 0.f, ts2 = 0.f;
#pragma unroll
        for (int i = 0; i < 8; ++i) { ts += wsum[i]; ts2 += wsum2[i]; }
        float mean = ts * (1.0f / 768.0f);
        float var  = ts2 * (1.0f / 768.0f) - mean * mean;
        mean_s = mean; rstd_s = rsqrtf(var + 1e-7f);
    }
    __syncthreads();
    const float mean = mean_s, rstd = rstd_s;
    for (int i = threadIdx.x; i < 768; i += 256) {
        float v = (xr[i] - mean) * rstd * gamma[i] + beta[i];
        out[(size_t)row * 768 + i] = f2bf(v);
    }
}

// ---------------------------------------------------------------------------
// bf16 WMMA GEMM:  C[M,N] = A[M,K] * B[K,N] (+bias, + epilogue)
// EPI: 0 = bf16 out (bias), 1 = bf16 out (bias + exact GELU),
//      2 = f32 out  (bias + residual add)
// Block: 256 threads (8 waves), 128x128 tile, K-step 32.
// Double-buffered LDS + register prefetch: next tile's global loads are in
// flight while the current tile's WMMAs execute; one barrier per K-step.
// ---------------------------------------------------------------------------
template <int EPI>
__global__ __launch_bounds__(256) void gemm_bf16_kernel(
    const unsigned short* __restrict__ A, const unsigned short* __restrict__ Bw,
    const float* __restrict__ bias, const float* __restrict__ resid,
    float* __restrict__ outF, unsigned short* __restrict__ outH,
    int M, int N, int K) {
    const int tid = threadIdx.x, lane = tid & 31, wave = tid >> 5;
    const int wm = wave & 3, wn = wave >> 2;       // 4 waves along M, 2 along N
    const int m0 = blockIdx.y * 128, n0 = blockIdx.x * 128;
    const int hi = lane >> 4, ln = lane & 15;

    __shared__ unsigned int Asm[2][128 * 17];       // 128 rows x 16 k-pairs (+pad)
    __shared__ unsigned int Bsm[2][32 * 65];        // 32 k-rows x 64 n-pairs (+pad)

    const unsigned int* Adw = (const unsigned int*)A;
    const unsigned int* Bdw = (const unsigned int*)Bw;
    const int Kd = K >> 1, Nd = N >> 1;

    unsigned int aReg[8], bReg[8];
    auto loadRegs = [&](int k0) {
#pragma unroll
        for (int t = 0; t < 8; ++t) {
            int e = tid + t * 256;
            aReg[t] = Adw[(m0 + (e >> 4)) * Kd + (k0 >> 1) + (e & 15)];
            bReg[t] = Bdw[(k0 + (e >> 6)) * Nd + (n0 >> 1) + (e & 63)];
        }
    };
    auto storeRegs = [&](int buf) {
#pragma unroll
        for (int t = 0; t < 8; ++t) {
            int e = tid + t * 256;
            Asm[buf][(e >> 4) * 17 + (e & 15)] = aReg[t];
            Bsm[buf][(e >> 6) * 65 + (e & 63)] = bReg[t];
        }
    };

    v8f zero = {0.f, 0.f, 0.f, 0.f, 0.f, 0.f, 0.f, 0.f};
    v8f acc[8];
#pragma unroll
    for (int i = 0; i < 8; ++i) acc[i] = zero;

    const int ktiles = K >> 5;
    loadRegs(0);
    storeRegs(0);
    __syncthreads();

    for (int kt = 0; kt < ktiles; ++kt) {
        const int buf = kt & 1;
        if (kt + 1 < ktiles) loadRegs((kt + 1) << 5);   // prefetch next tile

        v16bf af0 = load_fragA(Asm[buf], wm * 32,      17, 0, lane);
        v16bf af1 = load_fragA(Asm[buf], wm * 32 + 16, 17, 0, lane);
#pragma unroll
        for (int nf = 0; nf < 4; ++nf) {
            v16bf bfr = load_fragB(Bsm[buf], 0, 65, wn * 32 + nf * 8, lane);
            acc[nf]     = wmma_bf16(af0, bfr, acc[nf]);
            acc[4 + nf] = wmma_bf16(af1, bfr, acc[4 + nf]);
        }

        if (kt + 1 < ktiles) storeRegs(buf ^ 1);        // fill other buffer
        __syncthreads();
    }

#pragma unroll
    for (int mf = 0; mf < 2; ++mf)
#pragma unroll
        for (int nf = 0; nf < 4; ++nf) {
            const int col = n0 + wn * 64 + nf * 16 + ln;
            const float bcol = bias[col];
            v8f a = acc[mf * 4 + nf];
#pragma unroll
            for (int r = 0; r < 8; ++r) {
                const int row = m0 + wm * 32 + mf * 16 + r + 8 * hi;
                float v = a[r] + bcol;
                if constexpr (EPI == 1) v = gelu_exact(v);
                if constexpr (EPI == 2)
                    outF[(size_t)row * N + col] = v + resid[(size_t)row * N + col];
                else
                    outH[(size_t)row * N + col] = f2bf(v);
            }
        }
}

// ---------------------------------------------------------------------------
// Flash attention forward (causal, DK=64, H=12).
// qkv: packed per-token [Q(768)|K(768)|V(768)] bf16; ktg: K pre-transposed to
// [bh][d][s] bf16 so both K and V tiles stage with coalesced b32 copies.
// Block: 128 threads (4 waves) = one (b,h) and 64 query rows (16 per wave).
// ---------------------------------------------------------------------------
__global__ __launch_bounds__(128) void flash_attn_kernel(
    const unsigned short* __restrict__ qkv, const unsigned short* __restrict__ ktg,
    unsigned short* __restrict__ attn, int S) {
    const int b = blockIdx.x / 12, h = blockIdx.x % 12;
    const int q0 = blockIdx.y * 64;
    const int tid = threadIdx.x, wave = tid >> 5, lane = tid & 31;
    const int hi = lane >> 4, ln = lane & 15;
    const int bS = b * S;
    const unsigned int* qdw  = (const unsigned int*)qkv;
    const unsigned int* ktdw = (const unsigned int*)ktg;
    const int RS = 1152;                             // qkv row stride in dwords
    const int Sd = S >> 1;                           // ktg row stride in dwords
    const int kbase = blockIdx.x * 64;               // (b*12+h)*64 d-rows in ktg

    __shared__ unsigned int Kt[64 * 33];             // [d][kv-pairs]
    __shared__ unsigned int Vt[64 * 33];             // [kv][d-pairs]
    __shared__ unsigned int Pst[4 * 16 * 33];        // per-wave P staging [m][kv-pairs]
    unsigned int* Pw = Pst + wave * (16 * 33);
    unsigned short* Pw16 = (unsigned short*)Pw;

    // Q fragments (A-layout) straight from global, kept for the whole pass.
    const v16bf qa0 = load_fragA(qdw, bS + q0 + wave * 16, RS, h * 32,      lane);
    const v16bf qa1 = load_fragA(qdw, bS + q0 + wave * 16, RS, h * 32 + 16, lane);

    v8f zero = {0.f, 0.f, 0.f, 0.f, 0.f, 0.f, 0.f, 0.f};
    v8f o_acc[4];
#pragma unroll
    for (int d = 0; d < 4; ++d) o_acc[d] = zero;
    float m_i[8], l_i[8];
#pragma unroll
    for (int r = 0; r < 8; ++r) { m_i[r] = -1e30f; l_i[r] = 0.f; }

    const int jend = blockIdx.y;                     // causal: kv tiles 0..blockIdx.y
    for (int j = 0; j <= jend; ++j) {
        // ---- stage V tile: direct coalesced dword copy ----
#pragma unroll
        for (int t = 0; t < 16; ++t) {
            int e = tid + t * 128, r = e >> 5, c = e & 31;
            Vt[r * 33 + c] = qdw[(size_t)(bS + j * 64 + r) * RS + 768 + h * 32 + c];
        }
        // ---- stage K^T tile from ktg: direct coalesced dword copy ----
#pragma unroll
        for (int t = 0; t < 16; ++t) {
            int e = tid + t * 128, d = e >> 5, c = e & 31;
            Kt[d * 33 + c] = ktdw[(size_t)(kbase + d) * Sd + j * 32 + c];
        }
        __syncthreads();

        // ---- S = Q K^T ----
        v8f s_acc[4];
#pragma unroll
        for (int kvf = 0; kvf < 4; ++kvf) {
            s_acc[kvf] = zero;
            v16bf kb = load_fragB(Kt, 0,  33, kvf * 8, lane);
            s_acc[kvf] = wmma_bf16(qa0, kb, s_acc[kvf]);
            kb = load_fragB(Kt, 32, 33, kvf * 8, lane);
            s_acc[kvf] = wmma_bf16(qa1, kb, s_acc[kvf]);
        }

        // ---- online softmax (rows live in half-waves) ----
        const bool maskT = (j == jend);
#pragma unroll
        for (int r = 0; r < 8; ++r) {
            const int qrow = q0 + wave * 16 + r + 8 * hi;
            float sv[4];
#pragma unroll
            for (int kvf = 0; kvf < 4; ++kvf) {
                float s = s_acc[kvf][r] * 0.125f;                 // 1/sqrt(64)
                if (maskT) {
                    int kc = j * 64 + kvf * 16 + ln;
                    if (kc > qrow) s = -1e30f;
                }
                sv[kvf] = s;
            }
            float mrow = fmaxf(fmaxf(sv[0], sv[1]), fmaxf(sv[2], sv[3]));
#pragma unroll
            for (int o = 1; o < 16; o <<= 1) mrow = fmaxf(mrow, __shfl_xor(mrow, o, 32));
            const float mnew = fmaxf(m_i[r], mrow);
            const float alpha = __expf(m_i[r] - mnew);
            float pv[4], psum = 0.f;
#pragma unroll
            for (int kvf = 0; kvf < 4; ++kvf) { pv[kvf] = __expf(sv[kvf] - mnew); psum += pv[kvf]; }
#pragma unroll
            for (int o = 1; o < 16; o <<= 1) psum += __shfl_xor(psum, o, 32);
            l_i[r] = l_i[r] * alpha + psum;
            m_i[r] = mnew;
#pragma unroll
            for (int d = 0; d < 4; ++d) o_acc[d][r] *= alpha;
#pragma unroll
            for (int kvf = 0; kvf < 4; ++kvf)
                Pw16[(r + 8 * hi) * 66 + kvf * 16 + ln] = f2bf(pv[kvf]);
        }

        // ---- O += P V  (same-wave LDS: DS ops are in-order per wave) ----
#pragma unroll
        for (int ks = 0; ks < 2; ++ks) {
            v16bf pa = load_fragA(Pw, 0, 33, ks * 16, lane);
#pragma unroll
            for (int df = 0; df < 4; ++df) {
                v16bf vb = load_fragB(Vt, ks * 32, 33, df * 8, lane);
                o_acc[df] = wmma_bf16(pa, vb, o_acc[df]);
            }
        }
        __syncthreads();
    }

    // ---- normalize and write bf16 ----
#pragma unroll
    for (int r = 0; r < 8; ++r) {
        const float inv = 1.0f / l_i[r];
        const int row = q0 + wave * 16 + r + 8 * hi;
#pragma unroll
        for (int df = 0; df < 4; ++df) {
            const int col = h * 64 + df * 16 + ln;
            attn[(size_t)(bS + row) * 768 + col] = f2bf(o_acc[df][r] * inv);
        }
    }
}

// ---------------------------------------------------------------------------
// Host-side orchestration
// ---------------------------------------------------------------------------
extern "C" void kernel_launch(void* const* d_in, const int* in_sizes, int n_in,
                              void* d_out, int out_size, void* d_ws, size_t ws_size,
                              hipStream_t stream) {
    const float* x    = (const float*)d_in[0];
    const float* ln1g = (const float*)d_in[1];
    const float* ln1b = (const float*)d_in[2];
    const float* wq   = (const float*)d_in[3];
    const float* bq   = (const float*)d_in[4];
    const float* wk   = (const float*)d_in[5];
    const float* bk   = (const float*)d_in[6];
    const float* wv   = (const float*)d_in[7];
    const float* bv   = (const float*)d_in[8];
    const float* wo   = (const float*)d_in[9];
    const float* bo   = (const float*)d_in[10];
    const float* ln2g = (const float*)d_in[11];
    const float* ln2b = (const float*)d_in[12];
    const float* w1   = (const float*)d_in[13];
    const float* b1   = (const float*)d_in[14];
    const float* w2   = (const float*)d_in[15];
    const float* b2   = (const float*)d_in[16];
    float* out = (float*)d_out;

    constexpr int Bb = 2, S = 2048, D = 768, Hh = 12, DFF = 3072;
    const int Mtok = Bb * S;                         // 4096 token rows

    char* p = (char*)d_ws;
    auto alloc = [&](size_t bytes) -> void* {
        void* r = (void*)p; p += (bytes + 255) & ~(size_t)255; return r;
    };
    unsigned short* wqkv_bf = (unsigned short*)alloc((size_t)D * 3 * D * 2);
    unsigned short* wo_bf   = (unsigned short*)alloc((size_t)D * D * 2);
    unsigned short* w1_bf   = (unsigned short*)alloc((size_t)D * DFF * 2);
    unsigned short* w2_bf   = (unsigned short*)alloc((size_t)DFF * D * 2);
    float*          bqkv    = (float*)alloc((size_t)3 * D * 4);
    unsigned short* h1      = (unsigned short*)alloc((size_t)Mtok * D * 2);
    unsigned short* qkv     = (unsigned short*)alloc((size_t)Mtok * 3 * D * 2);
    unsigned short* ktg     = (unsigned short*)alloc((size_t)Mtok * D * 2);
    unsigned short* attn    = (unsigned short*)alloc((size_t)Mtok * D * 2);
    float*          x2      = (float*)alloc((size_t)Mtok * D * 4);
    unsigned short* h2      = (unsigned short*)alloc((size_t)Mtok * D * 2);
    unsigned short* ffh     = (unsigned short*)alloc((size_t)Mtok * DFF * 2);

    // Weight / bias conversion to bf16 (once per launch; deterministic).
    const int tw = D * D, t1 = D * DFF;
    convert_mat_kernel<<<(tw + 255) / 256, 256, 0, stream>>>(wq, wqkv_bf, tw, D, 3 * D, 0);
    convert_mat_kernel<<<(tw + 255) / 256, 256, 0, stream>>>(wk, wqkv_bf, tw, D, 3 * D, D);
    convert_mat_kernel<<<(tw + 255) / 256, 256, 0, stream>>>(wv, wqkv_bf, tw, D, 3 * D, 2 * D);
    convert_mat_kernel<<<(tw + 255) / 256, 256, 0, stream>>>(wo, wo_bf, tw, D, D, 0);
    convert_mat_kernel<<<(t1 + 255) / 256, 256, 0, stream>>>(w1, w1_bf, t1, DFF, DFF, 0);
    convert_mat_kernel<<<(t1 + 255) / 256, 256, 0, stream>>>(w2, w2_bf, t1, D, D, 0);
    concat_bias_kernel<<<3, 256, 0, stream>>>(bq, bk, bv, bqkv);

    // x -> LN1 -> h1 (bf16)
    ln_bf16_kernel<<<Mtok, 256, 0, stream>>>(x, ln1g, ln1b, h1);

    // QKV projection: [4096x768] x [768x2304] + bias -> bf16
    gemm_bf16_kernel<0><<<dim3(3 * D / 128, Mtok / 128), 256, 0, stream>>>(
        h1, wqkv_bf, bqkv, nullptr, nullptr, qkv, Mtok, 3 * D, D);

    // K rearrange once: [bh][d][s] for coalesced flash staging
    {
        const int total = Bb * Hh * 64 * S;
        k_transpose_kernel<<<(total + 255) / 256, 256, 0, stream>>>(qkv, ktg, S);
    }

    // Causal flash attention -> attn (bf16, [4096x768])
    flash_attn_kernel<<<dim3(Bb * Hh, S / 64), 128, 0, stream>>>(qkv, ktg, attn, S);

    // O-proj + residual: x2 = attn @ wo + bo + x   (f32)
    gemm_bf16_kernel<2><<<dim3(D / 128, Mtok / 128), 256, 0, stream>>>(
        attn, wo_bf, bo, x, x2, nullptr, Mtok, D, D);

    // x2 -> LN2 -> h2 (bf16)
    ln_bf16_kernel<<<Mtok, 256, 0, stream>>>(x2, ln2g, ln2b, h2);

    // FFN1 + exact GELU -> ffh (bf16)
    gemm_bf16_kernel<1><<<dim3(DFF / 128, Mtok / 128), 256, 0, stream>>>(
        h2, w1_bf, b1, nullptr, nullptr, ffh, Mtok, DFF, D);

    // FFN2 + residual -> out (f32)
    gemm_bf16_kernel<2><<<dim3(D / 128, Mtok / 128), 256, 0, stream>>>(
        ffh, w2_bf, b2, x2, out, nullptr, Mtok, D, DFF);

    (void)in_sizes; (void)n_in; (void)out_size; (void)ws_size;
}
